// JTNNEncoder_64836826301013
// MI455X (gfx1250) — matrix operand order repaired and checked by reading, not compiled, not run on hardware
//
#include <hip/hip_runtime.h>
#include <hip/hip_bf16.h>

typedef __attribute__((ext_vector_type(16))) __bf16 v16bf;
typedef __attribute__((ext_vector_type(8)))  __bf16 v8bf;
typedef __attribute__((ext_vector_type(4)))  __bf16 v4bf;
typedef __attribute__((ext_vector_type(8)))  float  v8f;
typedef __attribute__((ext_vector_type(4)))  unsigned int u32x4;
typedef __attribute__((ext_vector_type(8)))  unsigned int u32x8;

constexpr int HD = 256;   // hidden size
constexpr int KN = 6;     // padded neighbor count

// ---------------------------------------------------------------------------
// Weight f32 -> bf16 (with optional column offset to split concat weights)
__global__ __launch_bounds__(256) void k_cvtw(const float* __restrict__ src,
                                              __bf16* __restrict__ dst,
                                              int srcK, int off) {
  int r = blockIdx.x, c = threadIdx.x;
  dst[(size_t)r * HD + c] = (__bf16)src[(size_t)r * srcK + off + c];
}

// ---------------------------------------------------------------------------
// x[m] = emb[fnode[fmess[m]]]  -> bf16    (thread = (m, 8 cols))
__global__ __launch_bounds__(256) void k_embed(const int* __restrict__ fnode,
                                               const int* __restrict__ fmess,
                                               const float* __restrict__ emb,
                                               __bf16* __restrict__ xb, int M) {
  size_t t = (size_t)blockIdx.x * 256 + threadIdx.x;
  int m = (int)(t >> 5);
  if (m >= M) return;
  int c = ((int)t & 31) * 8;
  const float* e = emb + (size_t)fnode[fmess[m]] * HD + c;
  float4 e0 = *(const float4*)e;
  float4 e1 = *(const float4*)(e + 4);
  float ev[8] = {e0.x, e0.y, e0.z, e0.w, e1.x, e1.y, e1.z, e1.w};
  v8bf o;
#pragma unroll
  for (int i = 0; i < 8; ++i) o[i] = (__bf16)ev[i];
  *(v8bf*)(xb + (size_t)m * HD + c) = o;
}

// ---------------------------------------------------------------------------
__global__ __launch_bounds__(256) void k_zero(float* __restrict__ h,
                                              __bf16* __restrict__ hb, size_t n) {
  size_t i = (size_t)blockIdx.x * 256 + threadIdx.x;
  if (i < n) { h[i] = 0.f; hb[i] = (__bf16)0.f; }
}

// ---------------------------------------------------------------------------
// Per-message neighbor reduction + r-gate (thread = (m, 4 cols))
__global__ __launch_bounds__(256) void k_gather(const int* __restrict__ mg,
                                                const float* __restrict__ h,
                                                const __bf16* __restrict__ xWrb,
                                                const __bf16* __restrict__ hUrb,
                                                const float* __restrict__ bu,
                                                float* __restrict__ sumh,
                                                __bf16* __restrict__ sumhb,
                                                __bf16* __restrict__ sghb, int M) {
  size_t t = (size_t)blockIdx.x * 256 + threadIdx.x;
  int m = (int)(t >> 6);
  if (m >= M) return;
  int c = ((int)t & 63) * 4;
  size_t idx = (size_t)m * HD + c;

  v4bf xw4 = *(const v4bf*)(xWrb + idx);
  float4 b4 = *(const float4*)(bu + c);
  float xw[4] = {(float)xw4[0], (float)xw4[1], (float)xw4[2], (float)xw4[3]};
  float bb[4] = {b4.x, b4.y, b4.z, b4.w};
  float s[4] = {0.f, 0.f, 0.f, 0.f};
  float g[4] = {0.f, 0.f, 0.f, 0.f};

  const int* mgr = mg + (size_t)m * KN;
#pragma unroll
  for (int k = 0; k < KN; ++k) {
    int j = mgr[k];
    float4 hj = *(const float4*)(h + (size_t)j * HD + c);
    v4bf hu = *(const v4bf*)(hUrb + (size_t)j * HD + c);
    float hv[4] = {hj.x, hj.y, hj.z, hj.w};
#pragma unroll
    for (int i = 0; i < 4; ++i) {
      float pre = xw[i] + (float)hu[i] + bb[i];
      float r = 1.f / (1.f + __expf(-pre));
      s[i] += hv[i];
      g[i] += r * hv[i];
    }
  }
  *(float4*)(sumh + idx) = make_float4(s[0], s[1], s[2], s[3]);
  v4bf sb, gb;
#pragma unroll
  for (int i = 0; i < 4; ++i) { sb[i] = (__bf16)s[i]; gb[i] = (__bf16)g[i]; }
  *(v4bf*)(sumhb + idx) = sb;
  *(v4bf*)(sghb + idx) = gb;
}

// ---------------------------------------------------------------------------
// Readout gather for the B root nodes
__global__ __launch_bounds__(256) void k_cat(const int* __restrict__ scope,
                                             const int* __restrict__ fnode,
                                             const int* __restrict__ ngraph,
                                             const float* __restrict__ emb,
                                             const float* __restrict__ h,
                                             __bf16* __restrict__ catx,
                                             __bf16* __restrict__ cath) {
  int b = blockIdx.x, c = threadIdx.x;
  int node = scope[b];
  catx[(size_t)b * HD + c] = (__bf16)emb[(size_t)fnode[node] * HD + c];
  float s = 0.f;
  const int* ng = ngraph + (size_t)node * KN;
#pragma unroll
  for (int k = 0; k < KN; ++k) s += h[(size_t)ng[k] * HD + c];
  cath[(size_t)b * HD + c] = (__bf16)s;
}

// ---------------------------------------------------------------------------
// TDM: DMA the 256x256 bf16 weight matrix into LDS with a padded row pitch
// (512B rows + 16B pad = 528B) using one tensor_load_to_lds descriptor.
// D# layout per CDNA5 ISA. Issued by one wave per block.
__device__ __forceinline__ void tdm_load_weights(const __bf16* Bw) {
  unsigned long long ga = (unsigned long long)(const void*)Bw;
  u32x4 g0;
  g0[0] = 1u;                                          // count=1 (valid user D#)
  g0[1] = 0u;                                          // lds_addr = 0 (dyn-LDS base)
  g0[2] = (unsigned)(ga & 0xFFFFFFFFu);                // global_addr[31:0]
  g0[3] = (unsigned)((ga >> 32) & 0x01FFFFFFu) | (2u << 30);  // addr[56:32], type=2
  u32x8 g1;
  g1[0] = (3u << 16)      // data_size = 8B
        | (1u << 20)      // pad_enable
        | (6u << 22)      // pad_interval: 64 x 8B = 512B
        | (3u << 25);     // pad_amount: 4 DWORDs = 16B
  g1[1] = (64u << 16);    // tensor_dim0 = 64 (8B units)     [bits 79:48]
  g1[2] = (256u << 16);   // tensor_dim1 = 256               [bits 111:80]
  g1[3] = (64u << 16);    // tile_dim0 = 64                  [bits 127:112]
  g1[4] = 256u;           // tile_dim1 = 256, tile_dim2 = 0
  g1[5] = 64u;            // tensor_dim0_stride = 64         [bits 207:160]
  g1[6] = (16384u & 0xFFFFu) << 16;  // tensor_dim1_stride lo
  g1[7] = 16384u >> 16;              // tensor_dim1_stride hi
  asm volatile("tensor_load_to_lds %0, %1" :: "s"(g0), "s"(g1) : "memory");
  __builtin_amdgcn_s_wait_tensorcnt((short)0);
}

// ---------------------------------------------------------------------------
// WMMA GEMM: C[rows,256] = A[rows,256] x Bw[256,256]^T  (bf16 in, f32 accum)
// block = 8 waves; each wave owns a 64x64 output tile (16 accumulators) so each
// A/B fragment feeds 4 WMMAs. Block tile = 128 rows x 256 cols.
// ACT 0: outB = bf16(acc)
// ACT 1: outF = sigmoid(acc + addend + bias)                       (z gate)
// ACT 2: h = ((1-z)*sum_h + z*tanh(acc + addend + bias)) * mask    (GRU update)
// ACT 3: outF = relu(acc + addend + bias)                          (node_vecs)
template <int ACT>
__global__ __launch_bounds__(256) void k_gemm(const __bf16* __restrict__ A,
                                              const __bf16* __restrict__ Bw,
                                              const float* __restrict__ bias,
                                              const __bf16* __restrict__ addend,
                                              const float* __restrict__ zbuf,
                                              const float* __restrict__ sumh,
                                              float* __restrict__ outF,
                                              __bf16* __restrict__ outB, int rows) {
  constexpr int K = 256;
  constexpr int RP = K + 8;  // padded LDS row pitch (bf16 elems) -> conflict-free
  extern __shared__ __bf16 lb[];

  if (threadIdx.x < 32) tdm_load_weights(Bw);  // async 128KB weight stage via TDM
  __syncthreads();

  const int wave = threadIdx.x >> 5;
  const int lane = threadIdx.x & 31;
  const int lr = lane & 15, lh = lane >> 4;
  const int r0 = (blockIdx.x * 2 + (wave >> 2)) * 64;
  const int n0 = (wave & 3) * 64;

  const __bf16* Ap[4];
#pragma unroll
  for (int t = 0; t < 4; ++t) {
    int arow = r0 + t * 16 + lr;
    if (arow >= rows) arow = rows - 1;  // clamp; stores are guarded
    Ap[t] = A + (size_t)arow * K;
  }
  __builtin_prefetch(Ap[0], 0, 0);  // global_prefetch_b8

  const v8f zero = {0.f, 0.f, 0.f, 0.f, 0.f, 0.f, 0.f, 0.f};
  v8f acc[4][4];
#pragma unroll
  for (int t = 0; t < 4; ++t)
#pragma unroll
    for (int j = 0; j < 4; ++j) acc[t][j] = zero;

  for (int kk = 0; kk < K; kk += 32) {
    // A 16x32 bf16 fragments (ISA 7.12.2): lane lr row; K chunks [kk+lh*8,+8), +16
    union U { v16bf v; uint4 q[2]; };
    U a[4];
#pragma unroll
    for (int t = 0; t < 4; ++t) {
      a[t].q[0] = *(const uint4*)(Ap[t] + kk + lh * 8);
      a[t].q[1] = *(const uint4*)(Ap[t] + kk + 16 + lh * 8);
    }
#pragma unroll
    for (int j = 0; j < 4; ++j) {
      // B 32x16 fragment: lane holds column n, K chunks [kk+lh*8,+8), +16
      U b;
      const __bf16* bp = lb + (size_t)(n0 + j * 16 + lr) * RP + kk + lh * 8;
      b.q[0] = *(const uint4*)bp;
      b.q[1] = *(const uint4*)(bp + 16);
#pragma unroll
      for (int t = 0; t < 4; ++t)
        acc[t][j] = __builtin_amdgcn_wmma_f32_16x16x32_bf16(
            false, a[t].v, false, b.v, (short)0, acc[t][j], false, false);
    }
  }

#pragma unroll
  for (int t = 0; t < 4; ++t) {
#pragma unroll
    for (int j = 0; j < 4; ++j) {
#pragma unroll
      for (int i = 0; i < 8; ++i) {
        int row = r0 + t * 16 + lh * 8 + i;
        if (row >= rows) continue;
        int col = n0 + j * 16 + lr;
        size_t idx = (size_t)row * 256 + col;
        float v = acc[t][j][i];
        if constexpr (ACT == 0) {
          outB[idx] = (__bf16)v;
        } else if constexpr (ACT == 1) {
          v += (float)addend[idx] + bias[col];
          outF[idx] = 1.f / (1.f + __expf(-v));
        } else if constexpr (ACT == 2) {
          v += (float)addend[idx] + bias[col];
          float ph = tanhf(v);
          float zz = zbuf[idx];
          float hv = (1.f - zz) * sumh[idx] + zz * ph;
          if (row == 0) hv = 0.f;  // null-message mask
          outF[idx] = hv;
          outB[idx] = (__bf16)hv;
        } else {
          v += (float)addend[idx] + bias[col];
          outF[idx] = v > 0.f ? v : 0.f;
        }
      }
    }
  }
}

// ---------------------------------------------------------------------------
extern "C" void kernel_launch(void* const* d_in, const int* in_sizes, int n_in,
                              void* d_out, int out_size, void* d_ws, size_t ws_size,
                              hipStream_t stream) {
  const int*   fnode      = (const int*)d_in[0];
  const int*   fmess      = (const int*)d_in[1];
  const int*   node_graph = (const int*)d_in[2];
  const int*   mess_graph = (const int*)d_in[3];
  const int*   scope_st   = (const int*)d_in[4];
  const float* emb        = (const float*)d_in[5];
  const float* Wz = (const float*)d_in[6];
  const float* bz = (const float*)d_in[7];
  const float* Wr = (const float*)d_in[8];
  const float* Ur = (const float*)d_in[9];
  const float* bu = (const float*)d_in[10];
  const float* Wh = (const float*)d_in[11];
  const float* bh = (const float*)d_in[12];
  const float* Wo = (const float*)d_in[13];
  const float* bo = (const float*)d_in[14];

  const int M = in_sizes[1];
  const int B = in_sizes[4];
  const size_t MH = (size_t)M * HD;

  float* outTree = (float*)d_out;                    // [B, H]
  float* hF      = (float*)d_out + (size_t)B * HD;   // [M, H] final h lives in d_out

  char* ws = (char*)d_ws;
  size_t off = 0;
  auto alloc = [&](size_t bytes) -> char* {
    char* p = ws + off;
    off += (bytes + 255) & ~(size_t)255;
    return p;
  };
  __bf16* xb    = (__bf16*)alloc(MH * 2);
  __bf16* hb    = (__bf16*)alloc(MH * 2);
  __bf16* sumhb = (__bf16*)alloc(MH * 2);
  __bf16* sghb  = (__bf16*)alloc(MH * 2);
  __bf16* xWrb  = (__bf16*)alloc(MH * 2);
  __bf16* xWzb  = (__bf16*)alloc(MH * 2);
  __bf16* xWhb  = (__bf16*)alloc(MH * 2);
  __bf16* hUrb  = (__bf16*)alloc(MH * 2);
  float*  sumh  = (float*)alloc(MH * 4);
  float*  zbuf  = (float*)alloc(MH * 4);
  __bf16* Wrb  = (__bf16*)alloc((size_t)HD * HD * 2);
  __bf16* Urb  = (__bf16*)alloc((size_t)HD * HD * 2);
  __bf16* Wzxb = (__bf16*)alloc((size_t)HD * HD * 2);
  __bf16* Wzhb = (__bf16*)alloc((size_t)HD * HD * 2);
  __bf16* Whxb = (__bf16*)alloc((size_t)HD * HD * 2);
  __bf16* Whhb = (__bf16*)alloc((size_t)HD * HD * 2);
  __bf16* Woxb = (__bf16*)alloc((size_t)HD * HD * 2);
  __bf16* Wohb = (__bf16*)alloc((size_t)HD * HD * 2);
  __bf16* catx = (__bf16*)alloc((size_t)B * HD * 2);
  __bf16* cath = (__bf16*)alloc((size_t)B * HD * 2);
  __bf16* tmpb = (__bf16*)alloc((size_t)B * HD * 2);

  const size_t ldsB = (size_t)HD * (HD + 8) * 2;  // 135168 B padded weight stage

  // split + convert weights to bf16
  k_cvtw<<<HD, HD, 0, stream>>>(Wr, Wrb, HD, 0);
  k_cvtw<<<HD, HD, 0, stream>>>(Ur, Urb, HD, 0);
  k_cvtw<<<HD, HD, 0, stream>>>(Wz, Wzxb, 2 * HD, 0);
  k_cvtw<<<HD, HD, 0, stream>>>(Wz, Wzhb, 2 * HD, HD);
  k_cvtw<<<HD, HD, 0, stream>>>(Wh, Whxb, 2 * HD, 0);
  k_cvtw<<<HD, HD, 0, stream>>>(Wh, Whhb, 2 * HD, HD);
  k_cvtw<<<HD, HD, 0, stream>>>(Wo, Woxb, 2 * HD, 0);
  k_cvtw<<<HD, HD, 0, stream>>>(Wo, Wohb, 2 * HD, HD);

  // x = emb[fnode[fmess]] ; h = 0
  k_embed<<<(unsigned)(((size_t)M * 32 + 255) / 256), 256, 0, stream>>>(
      fnode, fmess, emb, xb, M);
  k_zero<<<(unsigned)((MH + 255) / 256), 256, 0, stream>>>(hF, hb, MH);

  const int gB = (M + 127) / 128;
  // loop-invariant x projections
  k_gemm<0><<<gB, 256, ldsB, stream>>>(xb, Wrb,  nullptr, nullptr, nullptr, nullptr, nullptr, xWrb, M);
  k_gemm<0><<<gB, 256, ldsB, stream>>>(xb, Wzxb, nullptr, nullptr, nullptr, nullptr, nullptr, xWzb, M);
  k_gemm<0><<<gB, 256, ldsB, stream>>>(xb, Whxb, nullptr, nullptr, nullptr, nullptr, nullptr, xWhb, M);

  const unsigned gatherB = (unsigned)(((size_t)M * 64 + 255) / 256);
  for (int d = 0; d < 6; ++d) {
    // hUr = h @ Ur^T  (hoisted out of the K-neighbor dimension)
    k_gemm<0><<<gB, 256, ldsB, stream>>>(hb, Urb, nullptr, nullptr, nullptr, nullptr, nullptr, hUrb, M);
    k_gather<<<gatherB, 256, 0, stream>>>(mess_graph, hF, xWrb, hUrb, bu,
                                          sumh, sumhb, sghb, M);
    // z = sigmoid(xWz + sum_h @ Wz_h^T + bz)
    k_gemm<1><<<gB, 256, ldsB, stream>>>(sumhb, Wzhb, bz, xWzb, nullptr, nullptr, zbuf, nullptr, M);
    // h = ((1-z)*sum_h + z*tanh(xWh + sgh @ Wh_h^T + bh)) * mask
    k_gemm<2><<<gB, 256, ldsB, stream>>>(sghb, Whhb, bh, xWhb, zbuf, sumh, hF, hb, M);
  }

  // readout: only the B=64 scope rows are needed
  k_cat<<<B, HD, 0, stream>>>(scope_st, fnode, node_graph, emb, hF, catx, cath);
  const int nb = (B + 127) / 128;
  k_gemm<0><<<nb, 256, ldsB, stream>>>(catx, Woxb, nullptr, nullptr, nullptr, nullptr, nullptr, tmpb, B);
  k_gemm<3><<<nb, 256, ldsB, stream>>>(cath, Wohb, bo, tmpb, nullptr, nullptr, outTree, nullptr, B);
}
